// DTFEModel_35966056136743
// MI455X (gfx1250) — matrix operand order repaired and checked by reading, not compiled, not run on hardware
//
#include <hip/hip_runtime.h>

#define NDEPTH 4
#define NHEADS 12
#define NDIM   768
#define NKEEP  358
#define NB     8
#define NTOK   512          // tokens per batch after embed (8*8*8)
#define KP     384          // padded kept tokens (24*16 = 12*32 = 6*64)
#define HD     64
#define KCONV  1536         // conv reduction 3*2*16*16
#define MTOK   (NB*NKEEP)   // 2864
#define MP     2880         // token rows padded to multiple of 64 (45*64)
#define MEMB   (NB*NTOK)    // 4096

typedef __bf16 bf16_t;
typedef __attribute__((ext_vector_type(16))) __bf16 v16bf;
typedef __attribute__((ext_vector_type(8)))  __bf16 v8bf;
typedef __attribute__((ext_vector_type(8)))  float  v8f;

// ---------------- WMMA tile loaders (contiguous k order, wave32) ----------------
// A (16x32, MxK): lane<16 -> K 0..15; lane>=16 -> K 16..31
__device__ __forceinline__ v16bf load_a_tile(const bf16_t* __restrict__ rowp, int sel) {
    return *reinterpret_cast<const v16bf*>(rowp + sel * 16);
}
// B (32x16, KxN) stored N-major (Bt[n*K+k]): lane<16 -> K 0..15; lane>=16 -> K 16..31
__device__ __forceinline__ v16bf load_b_tile(const bf16_t* __restrict__ colp, int sel) {
    return *reinterpret_cast<const v16bf*>(colp + sel * 16);
}

__device__ __forceinline__ float gelu_exact(float v) {
    return 0.5f * v * (1.0f + erff(v * 0.70710678118654752f));
}

// Register-blocked WMMA core. Loads all MT A-tiles per K-step, streams B-tiles
// one at a time to keep peak live VGPRs ~ (MT*NT*8 + MT*8 + 8) and avoid spills.
// Per K-step: MT*NT WMMAs on independent accumulators (no RAW hazard NOPs).
template <int MT, int NT>
__device__ __forceinline__ void wmma_core(const bf16_t* __restrict__ ap[MT],
                                          const bf16_t* __restrict__ bp[NT],
                                          int K, int sel, v8f acc[MT][NT]) {
    for (int k = 0; k < K; k += 32) {
        v16bf a[MT];
#pragma unroll
        for (int i = 0; i < MT; ++i) a[i] = load_a_tile(ap[i] + k, sel);
#pragma unroll
        for (int j = 0; j < NT; ++j) {
            v16bf b = load_b_tile(bp[j] + k, sel);
#pragma unroll
            for (int i = 0; i < MT; ++i)
                acc[i][j] = __builtin_amdgcn_wmma_f32_16x16x32_bf16(
                    false, a[i], false, b, (short)0, acc[i][j], false, false);
        }
    }
}

// ---------------- Generic bf16 WMMA GEMM: out = act(A @ Bt^T * alpha + bias + resid) ----
// A: (M x K) row-major bf16, Bt: (N x K) row-major bf16 (B transposed)
// one wave per (MT*16 x NT*16) block; grid = (N/(NT*16), Mpadded/(MT*16))
template <int MT, int NT, int ACT, bool RESID, bool OUTBF16>
__global__ void __launch_bounds__(32, 1)
gemm_bf16_wmma(const bf16_t* __restrict__ A,
               const bf16_t* __restrict__ Bt,
               const float*  __restrict__ bias,
               const float*  __restrict__ resid,
               void* __restrict__ outp,
               int N, int K, int Mvalid, float alpha) {
    const int lane = threadIdx.x;
    const int row  = lane & 15;
    const int sel  = lane >> 4;
    const int n0   = blockIdx.x * (NT * 16);
    const int m0   = blockIdx.y * (MT * 16);
    const bf16_t* ap[MT];
    const bf16_t* bp[NT];
#pragma unroll
    for (int i = 0; i < MT; ++i) ap[i] = A + (size_t)(m0 + i * 16 + row) * K;
#pragma unroll
    for (int j = 0; j < NT; ++j) bp[j] = Bt + (size_t)(n0 + j * 16 + row) * K;
    v8f acc[MT][NT];
#pragma unroll
    for (int i = 0; i < MT; ++i)
#pragma unroll
        for (int j = 0; j < NT; ++j) acc[i][j] = v8f{};
    wmma_core<MT, NT>(ap, bp, K, sel, acc);
#pragma unroll
    for (int i = 0; i < MT; ++i) {
#pragma unroll
        for (int j = 0; j < NT; ++j) {
            const int cc = n0 + j * 16 + row;
#pragma unroll
            for (int r = 0; r < 8; ++r) {
                const int rr = m0 + i * 16 + r + sel * 8;
                if (rr < Mvalid) {
                    float v = acc[i][j][r] * alpha;
                    if (bias) v += bias[cc];
                    if (RESID) v += resid[(size_t)rr * N + cc];
                    if (ACT == 1) v = gelu_exact(v);
                    if (OUTBF16) ((bf16_t*)outp)[(size_t)rr * N + cc] = (bf16_t)v;
                    else         ((float*)outp)[(size_t)rr * N + cc] = v;
                }
            }
        }
    }
}

// ---------------- Attention: scores = scale * Q @ K^T  (per b,h; padded KP) ----------
// q,k: (BH, KP, HD) bf16 ; attn: (BH, KP, KP) f32 ; 64x64 block per wave
__global__ void __launch_bounds__(32, 1)
attn_scores_wmma(const bf16_t* __restrict__ q,
                 const bf16_t* __restrict__ kmat,
                 float* __restrict__ attn, float scale) {
    const int bh = blockIdx.z;
    const int lane = threadIdx.x, row = lane & 15, sel = lane >> 4;
    const int n0 = blockIdx.x * 64, m0 = blockIdx.y * 64;
    const bf16_t* ap[4];
    const bf16_t* bp[4];
#pragma unroll
    for (int i = 0; i < 4; ++i) ap[i] = q    + (size_t)bh * KP * HD + (size_t)(m0 + i * 16 + row) * HD;
#pragma unroll
    for (int j = 0; j < 4; ++j) bp[j] = kmat + (size_t)bh * KP * HD + (size_t)(n0 + j * 16 + row) * HD;
    v8f acc[4][4];
#pragma unroll
    for (int i = 0; i < 4; ++i)
#pragma unroll
        for (int j = 0; j < 4; ++j) acc[i][j] = v8f{};
    wmma_core<4, 4>(ap, bp, HD, sel, acc);
    float* O = attn + (size_t)bh * KP * KP;
#pragma unroll
    for (int i = 0; i < 4; ++i)
#pragma unroll
        for (int j = 0; j < 4; ++j) {
            const int cc = n0 + j * 16 + row;
#pragma unroll
            for (int r = 0; r < 8; ++r)
                O[(size_t)(m0 + i * 16 + r + sel * 8) * KP + cc] = acc[i][j][r] * scale;
        }
}

// ---------------- Attention: O = P @ V  (A = probs bf16 (BH,KP,KP), Bt = vT (BH,HD,KP)) --
// 64x64 block per wave (N=HD=64 -> grid.x = 1); rows >= NKEEP dropped
__global__ void __launch_bounds__(32, 1)
attn_av_wmma(const bf16_t* __restrict__ probs,
             const bf16_t* __restrict__ vT,
             bf16_t* __restrict__ o_bf) {
    const int bh = blockIdx.z;
    const int b  = bh / NHEADS;
    const int h  = bh % NHEADS;
    const int lane = threadIdx.x, row = lane & 15, sel = lane >> 4;
    const int m0 = blockIdx.y * 64;
    const bf16_t* ap[4];
    const bf16_t* bp[4];
#pragma unroll
    for (int i = 0; i < 4; ++i) ap[i] = probs + (size_t)bh * KP * KP + (size_t)(m0 + i * 16 + row) * KP;
#pragma unroll
    for (int j = 0; j < 4; ++j) bp[j] = vT + (size_t)bh * HD * KP + (size_t)(j * 16 + row) * KP;
    v8f acc[4][4];
#pragma unroll
    for (int i = 0; i < 4; ++i)
#pragma unroll
        for (int j = 0; j < 4; ++j) acc[i][j] = v8f{};
    wmma_core<4, 4>(ap, bp, KP, sel, acc);
#pragma unroll
    for (int i = 0; i < 4; ++i)
#pragma unroll
        for (int j = 0; j < 4; ++j) {
            const int cc = j * 16 + row;
#pragma unroll
            for (int r = 0; r < 8; ++r) {
                const int rr = m0 + i * 16 + r + sel * 8;
                if (rr < NKEEP)
                    o_bf[(size_t)(b * NKEEP + rr) * NDIM + h * HD + cc] = (bf16_t)acc[i][j][r];
            }
        }
}

// ---------------- Elementwise / reduction helpers ------------------------------------
__global__ void convert_plain_kernel(const float* __restrict__ W, bf16_t* __restrict__ Wb, int n) {
    int i = blockIdx.x * blockDim.x + threadIdx.x;
    if (i < n) Wb[i] = (bf16_t)W[i];
}
// Wt[n*K + k] = W[k*N + n]
__global__ void convert_T_kernel(const float* __restrict__ W, bf16_t* __restrict__ Wt, int K, int N) {
    int i = blockIdx.x * blockDim.x + threadIdx.x;
    if (i >= K * N) return;
    int k = i / N, n = i % N;
    Wt[(size_t)n * K + k] = (bf16_t)W[i];
}
__global__ void fill_bf16_zero_kernel(bf16_t* __restrict__ p, int n) {
    int i = blockIdx.x * blockDim.x + threadIdx.x;
    if (i < n) p[i] = (bf16_t)0.f;
}

// patch matrix P (MEMB x KCONV) bf16 from x (NB,3,16,128,128)
__global__ void build_patches_kernel(const float* __restrict__ x, bf16_t* __restrict__ P) {
    int i = blockIdx.x * blockDim.x + threadIdx.x;
    if (i >= MEMB * KCONV) return;
    int m = i / KCONV, k = i % KCONV;
    int b = m >> 9, tok = m & 511;
    int d = tok >> 6, hh = (tok >> 3) & 7, w = tok & 7;
    int ci = k / 512, r = k % 512;
    int kd = r >> 8, kh = (r >> 4) & 15, kw = r & 15;
    size_t src = ((((size_t)b * 3 + ci) * 16 + (2 * d + kd)) * 128 + (16 * hh + kh)) * 128 + (16 * w + kw);
    P[i] = (bf16_t)x[src];
}

__global__ void motion_scores_kernel(const float* __restrict__ f, const float* __restrict__ fr,
                                     float* __restrict__ scores) {
    __shared__ float red[256];
    const int t = blockIdx.x;
    const float* a = f  + (size_t)t * NDIM;
    const float* b = fr + (size_t)t * NDIM;
    float s = 0.f;
    for (int c = threadIdx.x; c < NDIM; c += 256) {
        float d = 0.5f * (a[c] - b[c]);
        s += d * d;
    }
    red[threadIdx.x] = s; __syncthreads();
    for (int o = 128; o > 0; o >>= 1) {
        if (threadIdx.x < o) red[threadIdx.x] += red[threadIdx.x + o];
        __syncthreads();
    }
    if (threadIdx.x == 0) scores[t] = sqrtf(red[0]);
}

// exact stable top-k via rank counting (matches jax.lax.top_k ordering & tie-break)
__global__ void topk_kernel(const float* __restrict__ scores, int* __restrict__ idx) {
    __shared__ float s[NTOK];
    const int b = blockIdx.x, i = threadIdx.x;
    s[i] = scores[b * NTOK + i];
    __syncthreads();
    const float si = s[i];
    int rank = 0;
    for (int j = 0; j < NTOK; ++j) {
        float sj = s[j];
        rank += (sj > si) || (sj == si && j < i);
    }
    if (rank < NKEEP) idx[b * NKEEP + rank] = i;
}

__global__ void gather_tokens_kernel(const float* __restrict__ f, const int* __restrict__ idx,
                                     const float* __restrict__ policy_raw,
                                     float* __restrict__ h, float* __restrict__ pol) {
    const int t = blockIdx.x;          // 0..MTOK-1
    const int b = t / NKEEP;
    const int src = idx[t];
    const float* fr = f + ((size_t)b * NTOK + src) * NDIM;
    float* hr = h + (size_t)t * NDIM;
    for (int c = threadIdx.x; c < NDIM; c += 256) hr[c] = fr[c];
    if (threadIdx.x == 0) pol[t] = (policy_raw[t] > 0.f) ? 1.f : 0.f;
}

__global__ void layernorm_bf16_kernel(const float* __restrict__ x, const float* __restrict__ g,
                                      const float* __restrict__ be, bf16_t* __restrict__ y) {
    __shared__ float red[256];
    const int t = blockIdx.x;
    const float* xr = x + (size_t)t * NDIM;
    float s = 0.f;
    for (int c = threadIdx.x; c < NDIM; c += 256) s += xr[c];
    red[threadIdx.x] = s; __syncthreads();
    for (int o = 128; o > 0; o >>= 1) { if (threadIdx.x < o) red[threadIdx.x] += red[threadIdx.x + o]; __syncthreads(); }
    const float mean = red[0] / NDIM;
    __syncthreads();
    float s2 = 0.f;
    for (int c = threadIdx.x; c < NDIM; c += 256) { float d = xr[c] - mean; s2 += d * d; }
    red[threadIdx.x] = s2; __syncthreads();
    for (int o = 128; o > 0; o >>= 1) { if (threadIdx.x < o) red[threadIdx.x] += red[threadIdx.x + o]; __syncthreads(); }
    const float inv = rsqrtf(red[0] / NDIM + 1e-5f);
    bf16_t* yr = y + (size_t)t * NDIM;
    for (int c = threadIdx.x; c < NDIM; c += 256)
        yr[c] = (bf16_t)((xr[c] - mean) * inv * g[c] + be[c]);
}

// qkv (MP x 3*NDIM) f32 -> q,k (BH,KP,HD) bf16, vT (BH,HD,KP) bf16; zero padding
__global__ void repack_qkv_kernel(const float* __restrict__ qkv,
                                  bf16_t* __restrict__ qb, bf16_t* __restrict__ kb,
                                  bf16_t* __restrict__ vT) {
    int i = blockIdx.x * blockDim.x + threadIdx.x;
    const int tot = NB * NHEADS * KP * HD;
    if (i >= tot) return;
    int d = i % HD;
    int t = (i / HD) % KP;
    int bh = i / (HD * KP);
    int b = bh / NHEADS, h = bh % NHEADS;
    float qv = 0.f, kv = 0.f, vv = 0.f;
    if (t < NKEEP) {
        size_t base = (size_t)(b * NKEEP + t) * (3 * NDIM) + h * HD + d;
        qv = qkv[base];
        kv = qkv[base + NDIM];
        vv = qkv[base + 2 * NDIM];
    }
    qb[(size_t)bh * KP * HD + (size_t)t * HD + d] = (bf16_t)qv;
    kb[(size_t)bh * KP * HD + (size_t)t * HD + d] = (bf16_t)kv;
    vT[(size_t)bh * HD * KP + (size_t)d * KP + t] = (bf16_t)vv;
}

// policy-masked softmax: probs bf16 (BH,KP,KP); rows/cols >= NKEEP zeroed
__global__ void softmax_policy_kernel(const float* __restrict__ attn, const float* __restrict__ pol,
                                      bf16_t* __restrict__ probs) {
    const int rowi = blockIdx.x;    // 0..KP-1
    const int bh   = blockIdx.y;    // 0..NB*NHEADS-1
    const int b    = bh / NHEADS;
    bf16_t* orow = probs + ((size_t)bh * KP + rowi) * KP;
    if (rowi >= NKEEP) {
        for (int j = threadIdx.x; j < KP; j += 128) orow[j] = (bf16_t)0.f;
        return;
    }
    __shared__ float red[128];
    __shared__ float ev[KP];
    const float* arow = attn + ((size_t)bh * KP + rowi) * KP;
    float mx = -3.0e38f;
    for (int j = threadIdx.x; j < NKEEP; j += 128) mx = fmaxf(mx, arow[j]);
    red[threadIdx.x] = mx; __syncthreads();
    for (int o = 64; o > 0; o >>= 1) { if (threadIdx.x < o) red[threadIdx.x] = fmaxf(red[threadIdx.x], red[threadIdx.x + o]); __syncthreads(); }
    mx = red[0];
    __syncthreads();
    float s = 0.f;
    for (int j = threadIdx.x; j < NKEEP; j += 128) {
        float ap = (j == rowi) ? 1.f : pol[b * NKEEP + j];
        float e = expf(arow[j] - mx) * ap;
        ev[j] = e;
        s += e;
    }
    red[threadIdx.x] = s; __syncthreads();
    for (int o = 64; o > 0; o >>= 1) { if (threadIdx.x < o) red[threadIdx.x] += red[threadIdx.x + o]; __syncthreads(); }
    const float inv = 1.f / (red[0] + 1e-6f);
    const float c0 = 1e-6f / (float)NKEEP;
    for (int j = threadIdx.x; j < KP; j += 128)
        orow[j] = (j < NKEEP) ? (bf16_t)((ev[j] + c0) * inv) : (bf16_t)0.f;
}

// -------------------------------------------------------------------------------------
extern "C" void kernel_launch(void* const* d_in, const int* in_sizes, int n_in,
                              void* d_out, int out_size, void* d_ws, size_t ws_size,
                              hipStream_t stream) {
    (void)in_sizes; (void)n_in; (void)out_size; (void)ws_size;
    const float* x          = (const float*)d_in[0];
    const float* x_rev      = (const float*)d_in[1];
    const float* policy_raw = (const float*)d_in[2];
    const float* conv_w     = (const float*)d_in[3];
    const float* conv_b     = (const float*)d_in[4];
    const float* ln1_g      = (const float*)d_in[5];
    const float* ln1_b      = (const float*)d_in[6];
    const float* w_qkv      = (const float*)d_in[7];
    const float* w_proj     = (const float*)d_in[8];
    const float* b_proj     = (const float*)d_in[9];
    const float* ln2_g      = (const float*)d_in[10];
    const float* ln2_b      = (const float*)d_in[11];
    const float* w_fc1      = (const float*)d_in[12];
    const float* b_fc1      = (const float*)d_in[13];
    const float* w_fc2      = (const float*)d_in[14];
    const float* b_fc2      = (const float*)d_in[15];

    char* base = (char*)d_ws;
    size_t off = 0;
    auto alloc = [&](size_t bytes) -> void* {
        void* p = base + off;
        off += (bytes + 255) & ~(size_t)255;
        return p;
    };

    bf16_t* P       = (bf16_t*)alloc((size_t)MEMB * KCONV * 2);
    float*  f       = (float*) alloc((size_t)MEMB * NDIM * 4);
    float*  fr      = (float*) alloc((size_t)MEMB * NDIM * 4);
    float*  scores  = (float*) alloc((size_t)MEMB * 4);
    int*    idx     = (int*)   alloc((size_t)MTOK * 4);
    float*  pol     = (float*) alloc((size_t)MTOK * 4);
    float*  h       = (float*) alloc((size_t)MP * NDIM * 4);
    bf16_t* ybf     = (bf16_t*)alloc((size_t)MP * NDIM * 2);
    float*  qkv     = (float*) alloc((size_t)MP * 3 * NDIM * 4);
    bf16_t* qb      = (bf16_t*)alloc((size_t)NB * NHEADS * KP * HD * 2);
    bf16_t* kb      = (bf16_t*)alloc((size_t)NB * NHEADS * KP * HD * 2);
    bf16_t* vT      = (bf16_t*)alloc((size_t)NB * NHEADS * HD * KP * 2);
    float*  attn    = (float*) alloc((size_t)NB * NHEADS * KP * KP * 4);
    bf16_t* probs   = (bf16_t*)alloc((size_t)NB * NHEADS * KP * KP * 2);
    bf16_t* obf     = (bf16_t*)alloc((size_t)MP * NDIM * 2);
    bf16_t* hidbf   = (bf16_t*)alloc((size_t)MP * 4 * NDIM * 2);
    bf16_t* convWb  = (bf16_t*)alloc((size_t)NDIM * KCONV * 2);
    bf16_t* wqkvT   = (bf16_t*)alloc((size_t)NDEPTH * 3 * NDIM * NDIM * 2);
    bf16_t* wprojT  = (bf16_t*)alloc((size_t)NDEPTH * NDIM * NDIM * 2);
    bf16_t* wfc1T   = (bf16_t*)alloc((size_t)NDEPTH * 4 * NDIM * NDIM * 2);
    bf16_t* wfc2T   = (bf16_t*)alloc((size_t)NDEPTH * NDIM * 4 * NDIM * 2);

    const int T = 256;
    auto cdiv = [](int a, int b) { return (a + b - 1) / b; };

    // ---- weight conversions (conv_w is already N-major (O,K)) ----
    convert_plain_kernel<<<cdiv(NDIM * KCONV, T), T, 0, stream>>>(conv_w, convWb, NDIM * KCONV);
    for (int L = 0; L < NDEPTH; ++L) {
        convert_T_kernel<<<cdiv(NDIM * 3 * NDIM, T), T, 0, stream>>>(
            w_qkv + (size_t)L * NDIM * 3 * NDIM, wqkvT + (size_t)L * 3 * NDIM * NDIM, NDIM, 3 * NDIM);
        convert_T_kernel<<<cdiv(NDIM * NDIM, T), T, 0, stream>>>(
            w_proj + (size_t)L * NDIM * NDIM, wprojT + (size_t)L * NDIM * NDIM, NDIM, NDIM);
        convert_T_kernel<<<cdiv(NDIM * 4 * NDIM, T), T, 0, stream>>>(
            w_fc1 + (size_t)L * NDIM * 4 * NDIM, wfc1T + (size_t)L * 4 * NDIM * NDIM, NDIM, 4 * NDIM);
        convert_T_kernel<<<cdiv(4 * NDIM * NDIM, T), T, 0, stream>>>(
            w_fc2 + (size_t)L * 4 * NDIM * NDIM, wfc2T + (size_t)L * NDIM * 4 * NDIM, 4 * NDIM, NDIM);
    }

    // ---- zero the padded token rows of GEMM-A inputs not rewritten per layer ----
    const int padElems = (MP - MTOK) * NDIM;   // 16 rows * 768
    fill_bf16_zero_kernel<<<cdiv(padElems, T), T, 0, stream>>>(ybf + (size_t)MTOK * NDIM, padElems);
    fill_bf16_zero_kernel<<<cdiv(padElems, T), T, 0, stream>>>(obf + (size_t)MTOK * NDIM, padElems);

    // ---- patch embed (GEMM, K=1536) for x and x_rev ----
    build_patches_kernel<<<cdiv(MEMB * KCONV, T), T, 0, stream>>>(x, P);
    gemm_bf16_wmma<4, 4, 0, false, false><<<dim3(NDIM / 64, MEMB / 64), 32, 0, stream>>>(
        P, convWb, conv_b, nullptr, f, NDIM, KCONV, MEMB, 1.f);
    build_patches_kernel<<<cdiv(MEMB * KCONV, T), T, 0, stream>>>(x_rev, P);
    gemm_bf16_wmma<4, 4, 0, false, false><<<dim3(NDIM / 64, MEMB / 64), 32, 0, stream>>>(
        P, convWb, conv_b, nullptr, fr, NDIM, KCONV, MEMB, 1.f);

    // ---- motion scores, top-k selection, gather ----
    motion_scores_kernel<<<MEMB, 256, 0, stream>>>(f, fr, scores);
    topk_kernel<<<NB, NTOK, 0, stream>>>(scores, idx);
    gather_tokens_kernel<<<MTOK, 256, 0, stream>>>(f, idx, policy_raw, h, pol);

    // ---- transformer layers ----
    const int BH = NB * NHEADS;
    for (int L = 0; L < NDEPTH; ++L) {
        layernorm_bf16_kernel<<<MTOK, 256, 0, stream>>>(h, ln1_g + L * NDIM, ln1_b + L * NDIM, ybf);
        gemm_bf16_wmma<4, 4, 0, false, false><<<dim3(3 * NDIM / 64, MP / 64), 32, 0, stream>>>(
            ybf, wqkvT + (size_t)L * 3 * NDIM * NDIM, nullptr, nullptr, qkv, 3 * NDIM, NDIM, MP, 1.f);
        repack_qkv_kernel<<<cdiv(BH * KP * HD, T), T, 0, stream>>>(qkv, qb, kb, vT);
        attn_scores_wmma<<<dim3(KP / 64, KP / 64, BH), 32, 0, stream>>>(qb, kb, attn, 0.125f);
        softmax_policy_kernel<<<dim3(KP, BH), 128, 0, stream>>>(attn, pol, probs);
        attn_av_wmma<<<dim3(1, KP / 64, BH), 32, 0, stream>>>(probs, vT, obf);
        gemm_bf16_wmma<4, 4, 0, true, false><<<dim3(NDIM / 64, MP / 64), 32, 0, stream>>>(
            obf, wprojT + (size_t)L * NDIM * NDIM, b_proj + L * NDIM, h, h, NDIM, NDIM, MP, 1.f);
        layernorm_bf16_kernel<<<MTOK, 256, 0, stream>>>(h, ln2_g + L * NDIM, ln2_b + L * NDIM, ybf);
        gemm_bf16_wmma<4, 4, 1, false, true><<<dim3(4 * NDIM / 64, MP / 64), 32, 0, stream>>>(
            ybf, wfc1T + (size_t)L * 4 * NDIM * NDIM, b_fc1 + L * 4 * NDIM, nullptr, hidbf, 4 * NDIM, NDIM, MP, 1.f);
        void* outp = (L == NDEPTH - 1) ? d_out : (void*)h;
        const int mv = (L == NDEPTH - 1) ? MTOK : MP;
        gemm_bf16_wmma<4, 4, 0, true, false><<<dim3(NDIM / 64, MP / 64), 32, 0, stream>>>(
            hidbf, wfc2T + (size_t)L * NDIM * 4 * NDIM, b_fc2 + L * NDIM, h, outp, NDIM, 4 * NDIM, mv, 1.f);
    }
}
